// MarginalHead_31164282699883
// MI455X (gfx1250) — matrix operand order repaired and checked by reading, not compiled, not run on hardware
//
#include <hip/hip_runtime.h>
#include <hip/hip_bf16.h>

// ---------------------------------------------------------------------------
// Problem constants (from reference): B=4096, F=8, V=1000, D=1024
// ---------------------------------------------------------------------------
#define BSZ   4096
#define FD    8
#define VSZ   1000
#define VPAD  1024
#define DSZ   1024
#define LN_EPS 1e-5f

#define AS1 __attribute__((address_space(1)))
#define AS3 __attribute__((address_space(3)))

#if defined(__has_builtin)
#if __has_builtin(__builtin_amdgcn_global_load_async_to_lds_b128)
#define USE_ASYNC_LDS 1
#endif
#endif
#ifndef USE_ASYNC_LDS
#define USE_ASYNC_LDS 0
#endif

typedef __attribute__((ext_vector_type(16))) __bf16 v16bf;
typedef __attribute__((ext_vector_type(8)))  float  v8f;
typedef __attribute__((ext_vector_type(4)))  int    v4i;

union Frag16 {            // 32-byte A/B fragment for wmma_f32_16x16x32_bf16
    v16bf v;
    uint4 u[2];
};

__device__ __forceinline__ unsigned short f2bf_rne(float f) {
    union { float f; unsigned int u; } c; c.f = f;
    unsigned int u = c.u;
    u += 0x7FFFu + ((u >> 16) & 1u);   // round-to-nearest-even
    return (unsigned short)(u >> 16);
}

__device__ __forceinline__ void wait_async0() {
#if USE_ASYNC_LDS
#if __has_builtin(__builtin_amdgcn_s_wait_asynccnt)
    __builtin_amdgcn_s_wait_asynccnt(0);
#else
    asm volatile("s_wait_asynccnt 0" ::: "memory");
#endif
#endif
}

// ---------------------------------------------------------------------------
// Kernel 1: pred_W (F,D,V) f32  ->  Wt (F, VPAD, D) bf16, zero pad v>=V.
// Column-major-in-d layout so each WMMA B fragment is 2 contiguous 16B loads.
// ---------------------------------------------------------------------------
__global__ __launch_bounds__(256)
void convert_w_kernel(const float* __restrict__ W, unsigned short* __restrict__ Wt) {
    int idx = blockIdx.x * 256 + threadIdx.x;      // over F*VPAD*D = 8Mi
    int d = idx & (DSZ - 1);
    int t = idx >> 10;
    int v = t & (VPAD - 1);
    int f = t >> 10;
    float x = 0.0f;
    if (v < VSZ) x = W[(f * DSZ + d) * VSZ + v];
    Wt[idx] = f2bf_rne(x);
}

// ---------------------------------------------------------------------------
// Kernel 2: fused gather + prefix-sum + LayerNorm + exact GELU -> h bf16.
// One wave32 per batch row b; lane owns 32 floats (8 x float4, interleaved).
// ---------------------------------------------------------------------------
__global__ __launch_bounds__(256)
void compute_h_kernel(const float* __restrict__ inp,
                      const int* __restrict__ feats,
                      const float* __restrict__ tables,
                      const float* __restrict__ gamma,
                      const float* __restrict__ beta,
                      unsigned short* __restrict__ hbf) {
    const int wave = threadIdx.x >> 5;
    const int lane = threadIdx.x & 31;
    const int b = blockIdx.x * 8 + wave;

    const float4* inp4 = (const float4*)(inp + (size_t)b * DSZ);
    const float4* g4p  = (const float4*)gamma;
    const float4* b4p  = (const float4*)beta;

    float4 acc[8], g4[8], bt4[8];
#pragma unroll
    for (int i = 0; i < 8; ++i) {
        int c = lane + 32 * i;          // float4 index within the row
        acc[i] = inp4[c];
        g4[i]  = g4p[c];
        bt4[i] = b4p[c];
    }

    const float inv_d = 1.0f / (float)DSZ;
    uint2* hrow_base = (uint2*)(hbf + (size_t)b * FD * DSZ);

    for (int f = 0; f < FD; ++f) {
        // ---- LayerNorm statistics over D=1024 ----
        float s = 0.f, s2 = 0.f;
#pragma unroll
        for (int i = 0; i < 8; ++i) {
            float4 a = acc[i];
            s  += a.x + a.y + a.z + a.w;
            s2 += a.x * a.x + a.y * a.y + a.z * a.z + a.w * a.w;
        }
#pragma unroll
        for (int off = 16; off >= 1; off >>= 1) {
            s  += __shfl_xor(s,  off, 32);
            s2 += __shfl_xor(s2, off, 32);
        }
        float mu  = s * inv_d;
        float var = fmaxf(s2 * inv_d - mu * mu, 0.0f);
        float rs  = rsqrtf(var + LN_EPS);

        // ---- normalize, affine, exact GELU, pack bf16, store ----
        uint2* hrow = hrow_base + (size_t)f * (DSZ / 4);
#pragma unroll
        for (int i = 0; i < 8; ++i) {
            float4 a = acc[i], g = g4[i], bb = bt4[i];
            float y0 = (a.x - mu) * rs * g.x + bb.x;
            float y1 = (a.y - mu) * rs * g.y + bb.y;
            float y2 = (a.z - mu) * rs * g.z + bb.z;
            float y3 = (a.w - mu) * rs * g.w + bb.w;
            y0 = 0.5f * y0 * (1.0f + erff(y0 * 0.70710678118654752f));
            y1 = 0.5f * y1 * (1.0f + erff(y1 * 0.70710678118654752f));
            y2 = 0.5f * y2 * (1.0f + erff(y2 * 0.70710678118654752f));
            y3 = 0.5f * y3 * (1.0f + erff(y3 * 0.70710678118654752f));
            uint2 pk;
            pk.x = (unsigned)f2bf_rne(y0) | ((unsigned)f2bf_rne(y1) << 16);
            pk.y = (unsigned)f2bf_rne(y2) | ((unsigned)f2bf_rne(y3) << 16);
            hrow[lane + 32 * i] = pk;
        }

        // ---- accumulate feature f embedding for the next prefix step ----
        if (f < FD - 1) {
            int idx = feats[b * FD + f];
            const float4* e4 = (const float4*)(tables + ((size_t)f * VSZ + idx) * DSZ);
#pragma unroll
            for (int i = 0; i < 8; ++i) {
                float4 e = e4[lane + 32 * i];
                acc[i].x += e.x; acc[i].y += e.y; acc[i].z += e.z; acc[i].w += e.w;
            }
        }
    }
}

// ---------------------------------------------------------------------------
// Kernel 3: batched GEMM logits[b,f,v] = h[b,f,:] . Wt[f,v,:] + bias[f,v]
// Block tile 128(M) x 128(N); 8 waves as 4(M) x 2(N); wave tile 32x64.
// A double-buffered in LDS via GLOBAL_LOAD_ASYNC_TO_LDS_B128 (ASYNCcnt),
// B loaded directly from v-major bf16 Wt into WMMA fragments.
// v_wmma_f32_16x16x32_bf16, f32 accumulation.
// ---------------------------------------------------------------------------
#define LDA 40          // bf16 elements per LDS A row (32 + 8 pad)

__global__ __launch_bounds__(256)
void gemm_kernel(const unsigned short* __restrict__ hbf,
                 const unsigned short* __restrict__ Wt,
                 const float* __restrict__ bias,
                 float* __restrict__ out) {
    __shared__ unsigned short Atile[2][128 * LDA];

    const int tid  = threadIdx.x;
    const int lane = tid & 31;
    const int wave = tid >> 5;
    const int mw   = wave & 3;        // 0..3  -> M offset mw*32
    const int nw   = wave >> 2;       // 0..1  -> N offset nw*64

    const int f  = blockIdx.z;
    const int m0 = blockIdx.y * 128;  // batch-row tile
    const int n0 = blockIdx.x * 128;  // vocab tile (within VPAD)

    const int lhalf = lane >> 4;          // 0 | 1
    const int l15   = lane & 15;

    // This thread's two 16B chunks of the 128x32 A tile (512 chunks total)
    const int rowA0 = tid >> 1;                    // chunks 0,1 -> rows tid/2
    const int segA0 = (tid & 1) * 2;               // segments {0,1} or {2,3}
    // global source: h row (m0+row), f fixed, K window [k0, k0+32)
    const unsigned short* gA0 = hbf + (((size_t)(m0 + rowA0)) * FD + f) * DSZ + segA0 * 8;

    // B column base pointers (v-major Wt): col -> contiguous D run
    const unsigned short* bcol[4];
#pragma unroll
    for (int nf = 0; nf < 4; ++nf) {
        int v = n0 + nw * 64 + nf * 16 + l15;
        bcol[nf] = Wt + ((size_t)f * VPAD + v) * DSZ + lhalf * 8; // K offset 8*half
    }

    // stage the 128x32 A tile for K window k0 into buffer `buf`
    auto stage_a = [&](int k0, int buf) {
#if USE_ASYNC_LDS
#pragma unroll
        for (int i = 0; i < 2; ++i) {
            AS1 v4i* g = (AS1 v4i*)(gA0 + k0 + i * 8);
            AS3 v4i* l = (AS3 v4i*)&Atile[buf][rowA0 * LDA + (segA0 + i) * 8];
            __builtin_amdgcn_global_load_async_to_lds_b128(g, l, 0, 0);
        }
#else
#pragma unroll
        for (int i = 0; i < 2; ++i) {
            uint4 d = *(const uint4*)(gA0 + k0 + i * 8);
            *(uint4*)&Atile[buf][rowA0 * LDA + (segA0 + i) * 8] = d;
        }
#endif
    };

    v8f c[2][4] = {};

    stage_a(0, 0);                       // prologue: tile 0 in flight
    int buf = 0;

    for (int k0 = 0; k0 < DSZ; k0 += 32) {
        wait_async0();                   // this wave's stage of `buf` done
        __syncthreads();                 // publish `buf` to the workgroup

        if (k0 + 32 < DSZ)               // overlap next tile copy with compute
            stage_a(k0 + 32, buf ^ 1);

        // ---- load A fragments from LDS (ISA 16-bit A layout) ----
        Frag16 a[2];
#pragma unroll
        for (int mf = 0; mf < 2; ++mf) {
            int row = mw * 32 + mf * 16 + l15;
            int kb  = lhalf * 8;                      // lanes<16: K 0..7/16..23
            const unsigned short* ap = &Atile[buf][row * LDA + kb];
            a[mf].u[0] = *(const uint4*)(ap);         // K kb..kb+7
            a[mf].u[1] = *(const uint4*)(ap + 16);    // K kb+16..kb+23
        }

        // ---- load B fragments from global (K kb..kb+7, kb+16..kb+23 per lane) ----
        Frag16 bfr[4];
#pragma unroll
        for (int nf = 0; nf < 4; ++nf) {
            const unsigned short* bp = bcol[nf] + k0;
            bfr[nf].u[0] = *(const uint4*)(bp);
            bfr[nf].u[1] = *(const uint4*)(bp + 16);
        }

        // ---- 8 WMMAs per K-step ----
#pragma unroll
        for (int mf = 0; mf < 2; ++mf)
#pragma unroll
            for (int nf = 0; nf < 4; ++nf)
                c[mf][nf] = __builtin_amdgcn_wmma_f32_16x16x32_bf16(
                    false, a[mf].v, false, bfr[nf].v,
                    (short)0, c[mf][nf], false, false);

        buf ^= 1;
    }

    // ---- epilogue: + bias, masked store (v < 1000) ----
#pragma unroll
    for (int nf = 0; nf < 4; ++nf) {
        int v = n0 + nw * 64 + nf * 16 + l15;
        if (v < VSZ) {
            float bv = bias[f * VSZ + v];
#pragma unroll
            for (int mf = 0; mf < 2; ++mf) {
                int rbase = m0 + mw * 32 + mf * 16 + lhalf * 8;
#pragma unroll
                for (int j = 0; j < 8; ++j) {
                    int row = rbase + j;                      // batch index b
                    out[((size_t)row * FD + f) * VSZ + v] = c[mf][nf][j] + bv;
                }
            }
        }
    }
}

// ---------------------------------------------------------------------------
// Host-side launcher
// ---------------------------------------------------------------------------
extern "C" void kernel_launch(void* const* d_in, const int* in_sizes, int n_in,
                              void* d_out, int out_size, void* d_ws, size_t ws_size,
                              hipStream_t stream) {
    const float* inp    = (const float*)d_in[0];   // (B, D)
    const int*   feats  = (const int*)  d_in[1];   // (B, F)
    const float* tables = (const float*)d_in[2];   // (F, V, D)
    const float* gamma  = (const float*)d_in[3];   // (D,)
    const float* beta   = (const float*)d_in[4];   // (D,)
    const float* predW  = (const float*)d_in[5];   // (F, D, V)
    const float* predB  = (const float*)d_in[6];   // (F, V)
    float* out = (float*)d_out;                    // (B, F, V)

    // workspace: h bf16 (B*F*D) then Wt bf16 (F*VPAD*D)
    unsigned short* hbf = (unsigned short*)d_ws;
    unsigned short* Wt  = hbf + (size_t)BSZ * FD * DSZ;

    // 1) convert/transpose weights to bf16 v-major, zero-padded
    convert_w_kernel<<<(FD * VPAD * DSZ) / 256, 256, 0, stream>>>(predW, Wt);

    // 2) fused gather + prefix-sum + LN + GELU -> h (bf16)
    compute_h_kernel<<<BSZ / 8, 256, 0, stream>>>(inp, feats, tables, gamma, beta, hbf);

    // 3) WMMA GEMM + bias
    dim3 grid(VPAD / 128, BSZ / 128, FD);
    gemm_kernel<<<grid, 256, 0, stream>>>(hbf, Wt, predB, out);
}
